// MCRNNVAE_34754875359779
// MI455X (gfx1250) — matrix operand order repaired and checked by reading, not compiled, not run on hardware
//
#include <hip/hip_runtime.h>
#include <hip/hip_bf16.h>
#include <math.h>

typedef __attribute__((ext_vector_type(16))) _Float16 v16h;
typedef __attribute__((ext_vector_type(8)))  _Float16 v8h;
typedef __attribute__((ext_vector_type(8)))  float    v8f;

#define C_ 3
#define T_ 100
#define B_ 512
#define F_ 128
#define HX_ 128
#define HZ_ 128
#define H_ 256
#define EH_ 128
#define L_ 64

#define BT 32            // batch rows per workgroup (2 m-tiles)
#define NWAVE 8          // 256 threads = 8 wave32
#define FRAG 512         // f16 elements per packed 32x16 B-fragment

// Packed f16 weight offsets inside d_ws (units: f16 elements).
// Fragment (kt,nt) of W[K][N]: element s = lane*16+e holds
//   W[32*kt + (lane<16?0:16) + e][16*nt + (lane&15)]
#define OFF_WX   0                         // 3 ch * (Kt=4  * Nt=8)
#define OFF_WE   (OFF_WX  + 3*4*8*FRAG)    // 3 ch * (Kt=12 * Nt=8)
#define OFF_WQM  (OFF_WE  + 3*12*8*FRAG)   // 3 ch * (Kt=4  * Nt=4)
#define OFF_WZ   (OFF_WQM + 3*4*4*FRAG)    // 1    * (Kt=2  * Nt=8)
#define OFF_WD   (OFF_WZ  + 2*8*FRAG)      // 3 ch * (Kt=12 * Nt=8)
#define OFF_WPM  (OFF_WD  + 3*12*8*FRAG)   // 3 ch * (Kt=4  * Nt=8)
#define OFF_WIH  (OFF_WPM + 3*4*8*FRAG)    // 1    * (Kt=8  * Nt=16)
#define OFF_WHH  (OFF_WIH + 8*16*FRAG)     // 1    * (Kt=8  * Nt=16) (contiguous after WIH)
#define WS_F16_TOTAL (OFF_WHH + 8*16*FRAG) // 557056 f16 = ~1.09 MB

// LDS layout (f16 units). Wih+Whh resident for all 100 steps.
// h ping-pongs between region R0 and region R1; the off region doubles as the
// x|e staging area (both dead by the time h_new is written there).
#define LDS_WIH  0                 // 65536  (128 KB)
#define LDS_WHH  65536             // 65536  (128 KB)
#define LDS_R0   131072            // 8192   (h when t even; x|e when t odd)
#define LDS_R1   139264            // 8192   (x|e when t even; h when t odd)
#define LDS_XP   147456            // 4096
#define LDS_ZP   151552            // 4096
#define LDS_Z    155648            // 32x64 = 2048
#define LDS_D    157696            // 4096
#define LDS_TOT  161792            // 323584 B (<= 320 KB/WGP)

// ---------------------------------------------------------------------------
// One-time weight pack: f32 row-major [C][K][N] -> f16 WMMA B-fragment order.
// grid = (Nt, Kt, C), block = 512 (one thread per fragment element).
// ---------------------------------------------------------------------------
__global__ void vrnn_pack_w(const float* __restrict__ src,
                            _Float16* __restrict__ dst, int K, int N) {
  const int nt = blockIdx.x, kt = blockIdx.y, c = blockIdx.z;
  const int Nt = gridDim.x, Kt = gridDim.y;
  const int s = threadIdx.x;           // 0..511
  const int lane = s >> 4, e = s & 15;
  const int k = kt * 32 + ((lane < 16) ? 0 : 16) + e;
  const int n = nt * 16 + (lane & 15);
  const size_t frag = ((size_t)c * Kt + kt) * Nt + nt;
  dst[frag * FRAG + s] = (_Float16)src[((size_t)c * K + k) * N + n];
}

// ---------------------------------------------------------------------------
// A-fragment (16x32 f16) from row-major f16 activations in LDS.
// ---------------------------------------------------------------------------
__device__ __forceinline__ v16h load_a(const _Float16* A, int ldA, int m0,
                                       int k0, int lane) {
  const _Float16* p = A + (size_t)(m0 + (lane & 15)) * ldA + k0 +
                      ((lane < 16) ? 0 : 16);
  v8h lo = *(const v8h*)p;
  v8h hi = *(const v8h*)(p + 8);
  v16h a;
#pragma unroll
  for (int i = 0; i < 8; ++i) { a[i] = lo[i]; a[i + 8] = hi[i]; }
  return a;
}

__device__ __forceinline__ v16h load_b(const _Float16* W, int frag, int lane) {
  return *(const v16h*)(W + (size_t)frag * FRAG + lane * 16);
}

// B-fragment-major accumulate: wave owns one nt column; each B fragment is
// loaded ONCE and reused across both m-tiles. Operands are staged in batches
// of up to 4 K-tiles into distinct local values (12 loads back-to-back), so
// the scheduler can use partial dscnt waits and overlap LDS latency with the
// WMMA chain instead of a full wait per WMMA.
template <int NKT>
__device__ __forceinline__ void acc_block(v8f acc[2], const _Float16* A,
                                          int ldA, const _Float16* W,
                                          int ktBase, int Nt, int nt,
                                          int lane) {
#pragma unroll
  for (int kb = 0; kb < NKT; kb += 4) {
    const int NB = (NKT - kb) < 4 ? (NKT - kb) : 4;
    v16h aF[8], bF[4];
#pragma unroll
    for (int i = 0; i < NB; ++i) {
      const int kt = kb + i;
      bF[i]         = load_b(W, (ktBase + kt) * Nt + nt, lane);
      aF[2 * i]     = load_a(A, ldA, 0,  kt * 32, lane);
      aF[2 * i + 1] = load_a(A, ldA, 16, kt * 32, lane);
    }
#pragma unroll
    for (int i = 0; i < NB; ++i) {
      acc[0] = __builtin_amdgcn_wmma_f32_16x16x32_f16(
          false, aF[2 * i], false, bF[i], (short)0, acc[0], false, false);
      acc[1] = __builtin_amdgcn_wmma_f32_16x16x32_f16(
          false, aF[2 * i + 1], false, bF[i], (short)0, acc[1], false, false);
    }
  }
}

// Store both m-tiles of column nt to LDS as f16 with bias.
__device__ __forceinline__ void store2_lds(_Float16* D, int ldD, int nt,
                                           const v8f acc[2],
                                           const float* bias, int lane) {
  const int lm = lane & 15, hi8 = (lane < 16) ? 0 : 8;
  const float bv = bias[nt * 16 + lm];
  const int col = nt * 16 + lm;
#pragma unroll
  for (int mt = 0; mt < 2; ++mt)
#pragma unroll
    for (int r = 0; r < 8; ++r)
      D[(mt * 16 + hi8 + r) * ldD + col] = (_Float16)(acc[mt][r] + bv);
}

// ---------------------------------------------------------------------------
// Persistent per-(channel, batch-slice) recurrent kernel.
// grid = C*(B/BT) = 48 blocks, block = 256 threads (8 waves).
// ---------------------------------------------------------------------------
__global__ __launch_bounds__(256, 1)
void vrnn_step_kernel(const float* __restrict__ x,
                      const _Float16* __restrict__ ws,
                      float* __restrict__ out,
                      const float* __restrict__ bx_all,
                      const float* __restrict__ be_all,
                      const float* __restrict__ bqm_all,
                      const float* __restrict__ bz,
                      const float* __restrict__ bd_all,
                      const float* __restrict__ bpm_all,
                      const float* __restrict__ bih,
                      const float* __restrict__ bhh,
                      const float* __restrict__ h0) {
  __shared__ __align__(16) _Float16 smem[LDS_TOT];
  _Float16* sWih = smem + LDS_WIH;
  _Float16* sWhh = smem + LDS_WHH;
  _Float16* regA = smem + LDS_R0;
  _Float16* regB = smem + LDS_R1;
  _Float16* sXP  = smem + LDS_XP;
  _Float16* sZP  = smem + LDS_ZP;
  _Float16* sZ   = smem + LDS_Z;
  _Float16* sD   = smem + LDS_D;

  const int tid  = threadIdx.x;
  const int lane = tid & 31;
  const int wave = tid >> 5;
  const int c    = blockIdx.x / (B_ / BT);
  const int row0 = (blockIdx.x % (B_ / BT)) * BT;

  const _Float16* Wx  = ws + OFF_WX  + (size_t)c * 4 * 8 * FRAG;
  const _Float16* We  = ws + OFF_WE  + (size_t)c * 12 * 8 * FRAG;
  const _Float16* Wqm = ws + OFF_WQM + (size_t)c * 4 * 4 * FRAG;
  const _Float16* Wz  = ws + OFF_WZ;
  const _Float16* Wd  = ws + OFF_WD  + (size_t)c * 12 * 8 * FRAG;
  const _Float16* Wpm = ws + OFF_WPM + (size_t)c * 4 * 8 * FRAG;

  const float* bx  = bx_all  + c * HX_;
  const float* be  = be_all  + c * EH_;
  const float* bqm = bqm_all + c * L_;
  const float* bd  = bd_all  + c * EH_;
  const float* bpm = bpm_all + c * F_;

  // One-time: Wih+Whh (contiguous 256 KB in ws) -> resident LDS copy.
  {
    const uint4* g = (const uint4*)(ws + OFF_WIH);
    uint4* l = (uint4*)smem;  // LDS_WIH == 0
    for (int i = tid; i < (2 * 8 * 16 * FRAG) / 8; i += 256) l[i] = g[i];
  }
  // h <- h0 (f32 -> f16), into region R0 (h location for t=0).
  {
    const float* h0p = h0 + ((size_t)c * B_ + row0) * H_;
    for (int i = tid; i < BT * H_; i += 256) regA[i] = (_Float16)h0p[i];
  }
  __syncthreads();

  const int lm  = lane & 15;
  const int hi8 = (lane < 16) ? 0 : 8;

  for (int t = 0; t < T_; ++t) {
    // Ping-pong: h in one region; x|e staging (and h_new) in the other.
    _Float16* sH  = (t & 1) ? regB : regA;
    _Float16* sXE = (t & 1) ? regA : regB;
    _Float16* sX  = sXE;              // [32][128]
    _Float16* sE  = sXE + BT * F_;    // [32][128]
    _Float16* sHN = sXE;              // [32][256], overwrites dead x|e

    const float* xt = x + (((size_t)c * T_ + t) * B_ + row0) * F_;
    // Stage x tile into LDS as f16 (coalesced)
    for (int i = tid; i < BT * F_; i += 256) sX[i] = (_Float16)xt[i];
    // Prefetch next step's x tile: 128 threads x one 128B line = 16 KB
    if (t + 1 < T_ && tid < 128) {
      const float* xn = xt + (size_t)B_ * F_;
      __builtin_prefetch(xn + tid * 32, 0, 1);
    }
    __syncthreads();

    // (1) x_phi = x @ Wx + bx            [32x128], K=128 ; nt = wave
    {
      v8f acc[2]; acc[0] = v8f{}; acc[1] = v8f{};
      acc_block<4>(acc, sX, F_, Wx, 0, 8, wave, lane);
      store2_lds(sXP, HX_, wave, acc, bx, lane);
    }
    __syncthreads();

    // (2) e = [x_phi, h] @ We + be       [32x128], split-K 128 + 256
    {
      v8f acc[2]; acc[0] = v8f{}; acc[1] = v8f{};
      acc_block<4>(acc, sXP, HX_, We, 0, 8, wave, lane);
      acc_block<8>(acc, sH,  H_,  We, 4, 8, wave, lane);
      store2_lds(sE, EH_, wave, acc, be, lane);
    }
    __syncthreads();

    // (3) z = e @ Wqm + bqm              [32x64], K=128 ; waves 0..3 only
    if (wave < 4) {
      v8f acc[2]; acc[0] = v8f{}; acc[1] = v8f{};
      acc_block<4>(acc, sE, EH_, Wqm, 0, 4, wave, lane);
      store2_lds(sZ, L_, wave, acc, bqm, lane);
    }
    __syncthreads();

    // (4) z_phi = z @ Wz + bz            [32x128], K=64
    {
      v8f acc[2]; acc[0] = v8f{}; acc[1] = v8f{};
      acc_block<2>(acc, sZ, L_, Wz, 0, 8, wave, lane);
      store2_lds(sZP, HZ_, wave, acc, bz, lane);
    }
    __syncthreads();

    // (5) d = [z_phi, h] @ Wd + bd       [32x128], split-K 128 + 256
    {
      v8f acc[2]; acc[0] = v8f{}; acc[1] = v8f{};
      acc_block<4>(acc, sZP, HZ_, Wd, 0, 8, wave, lane);
      acc_block<8>(acc, sH,  H_,  Wd, 4, 8, wave, lane);
      store2_lds(sD, EH_, wave, acc, bd, lane);
    }
    __syncthreads();

    // (6) mu = d @ Wpm + bpm -> global f32 output
    {
      v8f acc[2]; acc[0] = v8f{}; acc[1] = v8f{};
      acc_block<4>(acc, sD, EH_, Wpm, 0, 8, wave, lane);
      const float bv = bpm[wave * 16 + lm];
      const int col = wave * 16 + lm;
#pragma unroll
      for (int mt = 0; mt < 2; ++mt) {
        float* op = out + (((size_t)c * T_ + t) * B_ + row0 + mt * 16 + hi8) *
                        F_ + col;
#pragma unroll
        for (int r = 0; r < 8; ++r) op[(size_t)r * F_] = acc[mt][r] + bv;
      }
    }
    // No barrier: stage 7 reads sXP/sZP/sH and writes sHN (dead x|e region),
    // disjoint from stage 6's sD reads / global writes.

    // (7) h_new = tanh(x_phi@Wih[:128] + z_phi@Wih[128:] + h@Whh + bih + bhh)
    //     Wih/Whh B-fragments come from resident LDS. nt = wave, wave+8.
    //     Keep the half loop rolled to cap VGPR pressure.
#pragma unroll 1
    for (int half = 0; half < 2; ++half) {
      const int nt = wave + half * 8;
      v8f acc[2]; acc[0] = v8f{}; acc[1] = v8f{};
      acc_block<4>(acc, sXP, HX_, sWih, 0, 16, nt, lane);
      acc_block<4>(acc, sZP, HZ_, sWih, 4, 16, nt, lane);
      acc_block<8>(acc, sH,  H_,  sWhh, 0, 16, nt, lane);
      const float bv = bih[nt * 16 + lm] + bhh[nt * 16 + lm];
      const int col = nt * 16 + lm;
#pragma unroll
      for (int mt = 0; mt < 2; ++mt)
#pragma unroll
        for (int r = 0; r < 8; ++r)
          sHN[(mt * 16 + hi8 + r) * H_ + col] = (_Float16)tanhf(acc[mt][r] + bv);
    }
    // One barrier: stage-7 reads of old h must finish before next step stages
    // x into that region; also publishes h_new for next step.
    __syncthreads();
  }
}

extern "C" void kernel_launch(void* const* d_in, const int* in_sizes, int n_in,
                              void* d_out, int out_size, void* d_ws,
                              size_t ws_size, hipStream_t stream) {
  (void)in_sizes; (void)n_in; (void)out_size; (void)ws_size;

  const float* x = (const float*)d_in[0];
  _Float16* ws = (_Float16*)d_ws;

  const dim3 b512(512);
  // Pack all weights into WMMA B-fragment order (f16) once per launch.
  vrnn_pack_w<<<dim3(8, 4, 3),  b512, 0, stream>>>((const float*)d_in[1],  ws + OFF_WX,  128, 128);
  vrnn_pack_w<<<dim3(8, 12, 3), b512, 0, stream>>>((const float*)d_in[3],  ws + OFF_WE,  384, 128);
  vrnn_pack_w<<<dim3(4, 4, 3),  b512, 0, stream>>>((const float*)d_in[5],  ws + OFF_WQM, 128, 64);
  vrnn_pack_w<<<dim3(8, 2, 1),  b512, 0, stream>>>((const float*)d_in[7],  ws + OFF_WZ,  64,  128);
  vrnn_pack_w<<<dim3(8, 12, 3), b512, 0, stream>>>((const float*)d_in[9],  ws + OFF_WD,  384, 128);
  vrnn_pack_w<<<dim3(8, 4, 3),  b512, 0, stream>>>((const float*)d_in[11], ws + OFF_WPM, 128, 128);
  vrnn_pack_w<<<dim3(16, 8, 1), b512, 0, stream>>>((const float*)d_in[13], ws + OFF_WIH, 256, 256);
  vrnn_pack_w<<<dim3(16, 8, 1), b512, 0, stream>>>((const float*)d_in[14], ws + OFF_WHH, 256, 256);

  vrnn_step_kernel<<<dim3(C_ * (B_ / BT)), dim3(256), 0, stream>>>(
      x, ws, (float*)d_out,
      (const float*)d_in[2],  (const float*)d_in[4],  (const float*)d_in[6],
      (const float*)d_in[8],  (const float*)d_in[10], (const float*)d_in[12],
      (const float*)d_in[15], (const float*)d_in[16], (const float*)d_in[17]);
}
